// SwinGraphBlock_28097676050904
// MI455X (gfx1250) — matrix operand order repaired and checked by reading, not compiled, not run on hardware
//
#include <hip/hip_runtime.h>

typedef __attribute__((ext_vector_type(16))) _Float16 v16h;
typedef __attribute__((ext_vector_type(8)))  float    v8f;
typedef __attribute__((ext_vector_type(4)))  int      i32x4;

#define NN   4096
#define CC   128
#define KCL  32
#define MM   128
#define HH   8
#define DD   64
#define INNR 512
#define EE   65536

// ---------------- CDNA5 async global->LDS copy (guarded) ----------------

#if __has_builtin(__builtin_amdgcn_global_load_async_to_lds_b128)
#define HAVE_ASYNC_COPY 1
#else
#define HAVE_ASYNC_COPY 0
#endif

__device__ __forceinline__ void async_wait0() {
#if __has_builtin(__builtin_amdgcn_s_wait_asynccnt)
  __builtin_amdgcn_s_wait_asynccnt(0);
#elif HAVE_ASYNC_COPY
  asm volatile("s_wait_asynccnt 0x0" ::: "memory");
#endif
}

// Copy one 128x128 f32 tile from global into LDS. Uses the gfx1250
// GLOBAL_LOAD_ASYNC_TO_LDS_B128 path (ASYNCcnt-tracked, no VGPR round-trip).
// Builtin signature: (int4 addrspace(1)*, int4 addrspace(3)*, imm off, imm cpol).
__device__ __forceinline__ void tile_copy_in(const float* __restrict__ g,
                                             float* s, int tid) {
#if HAVE_ASYNC_COPY
  typedef __attribute__((address_space(1))) i32x4 as1_i32x4;
  typedef __attribute__((address_space(3))) i32x4 as3_i32x4;
  for (int i = tid * 4; i < MM * CC; i += 256 * 4)
    __builtin_amdgcn_global_load_async_to_lds_b128(
        (as1_i32x4*)(g + i), (as3_i32x4*)(s + i), 0, 0);
  async_wait0();
#else
  const float4* gx = (const float4*)g;
  float4* sx = (float4*)s;
  for (int i = tid; i < MM * CC / 4; i += 256) sx[i] = gx[i];
#endif
}

// ---------------- WMMA helpers (CDNA5 f16 -> f32, 16x16x32) ----------------

__device__ __forceinline__ v8f wmma16(v16h a, v16h b, v8f c) {
  // D = A(16x32 f16) * B(32x16 f16) + C(16x16 f32)
  return __builtin_amdgcn_wmma_f32_16x16x32_f16(false, a, false, b, (short)0, c,
                                                false, false);
}

// A fragment (16x32 f16), row-major source with leading dim `ld` (halves).
// Lane layout: lanes 0-15 row=lane, K=0..15; lanes 16-31
// row=lane-16, K=16..31 (contiguous, matches load_b).
__device__ __forceinline__ v16h load_a(const _Float16* base, int ld) {
  const int lane = threadIdx.x & 31;
  const _Float16* p = base + (lane & 15) * ld + ((lane >> 4) << 4);
  v16h f;
#pragma unroll
  for (int i = 0; i < 16; ++i) f[i] = p[i];
  return f;
}

// B fragment (32x16 f16) from a TRANSPOSED source Bt[n][k] with leading dim
// `ld` (halves): lane holds column n=lane%16, contiguous K run of 16.
__device__ __forceinline__ v16h load_b(const _Float16* base, int ld) {
  const int lane = threadIdx.x & 31;
  const _Float16* p = base + (lane & 15) * ld + ((lane >> 4) << 4);
  v16h f;
#pragma unroll
  for (int i = 0; i < 16; ++i) f[i] = p[i];
  return f;
}

__device__ __forceinline__ float wave_sum(float v) {
#pragma unroll
  for (int m = 16; m >= 1; m >>= 1) v += __shfl_xor(v, m, 32);
  return v;
}
__device__ __forceinline__ float wave_max(float v) {
#pragma unroll
  for (int m = 16; m >= 1; m >>= 1) v = fmaxf(v, __shfl_xor(v, m, 32));
  return v;
}

// ---------------- adjacency bitmask ----------------

__global__ void k_zero_adj(unsigned* adj) {
  int i = blockIdx.x * blockDim.x + threadIdx.x;
  if (i < KCL * MM * 4) adj[i] = 0u;
}

__global__ void k_build_adj(const int* __restrict__ edge, unsigned* __restrict__ adj) {
  int e = blockIdx.x * blockDim.x + threadIdx.x;
  if (e >= EE) return;
  int s = edge[e], d = edge[EE + e];
  int cs = s >> 7, cd = d >> 7;
  if (cs == cd) {
    int i = s & 127, j = d & 127;
    atomicOr(&adj[cs * (MM * 4) + i * 4 + (j >> 5)], 1u << (j & 31));
  }
}

// ---------------- weight prep: f32 -> f16 transposed Bt[n][k]; e0/e1 ----------------

__global__ void k_prep(const float* __restrict__ Wq, const float* __restrict__ Wkv,
                       const float* __restrict__ Wo, const float* __restrict__ W1,
                       const float* __restrict__ W2, const float* __restrict__ adj_emb,
                       const float* __restrict__ We, const float* __restrict__ be,
                       _Float16* WqT, _Float16* WkvT, _Float16* WoT,
                       _Float16* W1T, _Float16* W2T, float* e01) {
  const int total = 65536 + 131072 + 65536 + 65536 + 65536 + 1024;
  for (int i = blockIdx.x * blockDim.x + threadIdx.x; i < total;
       i += gridDim.x * blockDim.x) {
    int t = i;
    if (t < 65536) {                    // WqT [512][128]
      int n = t >> 7, k = t & 127;
      WqT[n * 128 + k] = (_Float16)Wq[k * INNR + n];
    } else if ((t -= 65536) < 131072) { // WkvT [1024][128]
      int n = t >> 7, k = t & 127;
      WkvT[n * 128 + k] = (_Float16)Wkv[k * (2 * INNR) + n];
    } else if ((t -= 131072) < 65536) { // WoT [128][512]
      int n = t >> 9, k = t & 511;
      WoT[n * 512 + k] = (_Float16)Wo[k * CC + n];
    } else if ((t -= 65536) < 65536) {  // W1T [512][128]
      int n = t >> 7, k = t & 127;
      W1T[n * 128 + k] = (_Float16)W1[k * (4 * CC) + n];
    } else if ((t -= 65536) < 65536) {  // W2T [128][512]
      int n = t >> 9, k = t & 511;
      W2T[n * 512 + k] = (_Float16)W2[k * CC + n];
    } else {                            // e01 [2][512]: adj_emb[b] @ We + be
      t -= 65536;
      int b = t >> 9, j = t & 511;
      float s = be[j];
      for (int c = 0; c < CC; ++c) s += adj_emb[b * CC + c] * We[c * INNR + j];
      e01[b * INNR + j] = s;
    }
  }
}

// ---------------- block 1: LN1 + edge-biased attention + Wo + gate1 ----------------

__global__ void __launch_bounds__(256)
k_block1(const float* __restrict__ x, const unsigned* __restrict__ adjw,
         const float* __restrict__ e01, const _Float16* __restrict__ WqT,
         const _Float16* __restrict__ WkvT, const _Float16* __restrict__ WoT,
         const float* __restrict__ bq, const float* __restrict__ bkv,
         const float* __restrict__ bo, const float* __restrict__ g1,
         const float* __restrict__ ln1g, const float* __restrict__ ln1b,
         float* __restrict__ nodes1) {
  __shared__ float    xres[MM][CC];        // residual (original nodes)     64 KB
  __shared__ _Float16 y16[MM][136];        // LN output (f16)               34 KB
  __shared__ _Float16 qh[MM][72];          // per-head q                    18 KB
  __shared__ _Float16 kh[MM][72];          // per-head k                    18 KB
  __shared__ _Float16 vt[DD][136];         // per-head v, TRANSPOSED [d][j] 17 KB
  __shared__ _Float16 oh[MM][72];          // per-head attention output     18 KB
  __shared__ float    simf[MM][CC];        // sim f32; att f16 packed; xn   64 KB
  __shared__ float    qe0s[MM], qe1s[MM], a1s[MM];

  const int tid = threadIdx.x, w = tid >> 5, lane = tid & 31;
  const int cl = blockIdx.x;
  const unsigned* adjc = adjw + cl * (MM * 4);

  if (tid < 128) __builtin_prefetch(WoT + (size_t)tid * 512, 0, 3);

  // ---- load cluster tile (contiguous rows) via async-to-LDS ----
  tile_copy_in(x + (size_t)cl * MM * CC, &xres[0][0], tid);
  __syncthreads();

  // ---- LayerNorm 1 (one wave per row, 4 cols/lane) ----
  for (int r = w; r < MM; r += 8) {
    const int c = lane * 4;
    float4 v = *(const float4*)&xres[r][c];
    float s  = wave_sum(v.x + v.y + v.z + v.w);
    float ss = wave_sum(v.x * v.x + v.y * v.y + v.z * v.z + v.w * v.w);
    float mu = s * (1.f / CC);
    float rs = rsqrtf(ss * (1.f / CC) - mu * mu + 1e-5f);
    y16[r][c + 0] = (_Float16)((v.x - mu) * rs * ln1g[c + 0] + ln1b[c + 0]);
    y16[r][c + 1] = (_Float16)((v.y - mu) * rs * ln1g[c + 1] + ln1b[c + 1]);
    y16[r][c + 2] = (_Float16)((v.z - mu) * rs * ln1g[c + 2] + ln1b[c + 2]);
    y16[r][c + 3] = (_Float16)((v.w - mu) * rs * ln1g[c + 3] + ln1b[c + 3]);
  }
  __syncthreads();

  // persistent Wo-projection accumulators: wave w owns output rows [16w,16w+16)
  v8f accP[8];
  {
    v8f z = {};
#pragma unroll
    for (int i = 0; i < 8; ++i) accP[i] = z;
  }

  for (int h = 0; h < HH; ++h) {
    const float* e0 = e01 + h * DD;
    const float* e1 = e01 + INNR + h * DD;

    // ---- q,k,v = y @ W (per-head 128x64 each); v stored transposed ----
#pragma unroll
    for (int t = 0; t < 4; ++t) {
      int idx = w * 4 + t, rb = idx >> 2, cb = idx & 3;
      v8f aq = {}, ak = {}, av = {};
#pragma unroll
      for (int kk = 0; kk < 4; ++kk) {
        v16h a = load_a(&y16[rb * 16][kk * 32], 136);
        v16h b = load_b(WqT + (size_t)(h * DD + cb * 16) * CC + kk * 32, CC);
        aq = wmma16(a, b, aq);
        b = load_b(WkvT + (size_t)(h * DD + cb * 16) * CC + kk * 32, CC);
        ak = wmma16(a, b, ak);
        b = load_b(WkvT + (size_t)(INNR + h * DD + cb * 16) * CC + kk * 32, CC);
        av = wmma16(a, b, av);
      }
      int dc = cb * 16 + (lane & 15);
      int r0 = rb * 16 + ((lane >> 4) << 3);
      float bqv = bq[h * DD + dc];
      float bkk = bkv[h * DD + dc];
      float bvv = bkv[INNR + h * DD + dc];
#pragma unroll
      for (int i = 0; i < 8; ++i) {
        qh[r0 + i][dc] = (_Float16)(aq[i] + bqv);
        kh[r0 + i][dc] = (_Float16)(ak[i] + bkk);
        vt[dc][r0 + i] = (_Float16)(av[i] + bvv);
      }
    }
    __syncthreads();

    // ---- rotary (interleaved pairs, theta=1e4) + qe0/qe1 row dots ----
    for (int r = w; r < MM; r += 8) {
      int d0 = lane * 2, d1 = d0 + 1;
      float inv = __powf(10000.f, -(float)lane / 32.f);  // 2p/D = lane/32
      float sn, cs;
      __sincosf((float)r * inv, &sn, &cs);
      float q0 = (float)qh[r][d0], q1 = (float)qh[r][d1];
      float nq0 = q0 * cs - q1 * sn, nq1 = q1 * cs + q0 * sn;
      qh[r][d0] = (_Float16)nq0;
      qh[r][d1] = (_Float16)nq1;
      float k0 = (float)kh[r][d0], k1 = (float)kh[r][d1];
      kh[r][d0] = (_Float16)(k0 * cs - k1 * sn);
      kh[r][d1] = (_Float16)(k1 * cs + k0 * sn);
      float p0 = wave_sum(nq0 * e0[d0] + nq1 * e0[d1]);
      float p1 = wave_sum(nq0 * e1[d0] + nq1 * e1[d1]);
      if (lane == 0) { qe0s[r] = p0; qe1s[r] = p1; }
    }
    __syncthreads();

    // ---- sim = (q k^T + qe[adj]) * D^-0.5 ----
#pragma unroll
    for (int t = 0; t < 8; ++t) {
      int idx = w * 8 + t, rb = idx >> 3, cb = idx & 7;
      v8f acc = {};
#pragma unroll
      for (int kk = 0; kk < 2; ++kk) {
        v16h a = load_a(&qh[rb * 16][kk * 32], 72);
        v16h b = load_b(&kh[cb * 16][kk * 32], 72);  // Bt[n=j][k=d] = k rows
        acc = wmma16(a, b, acc);
      }
      int jc = cb * 16 + (lane & 15);
      int r0 = rb * 16 + ((lane >> 4) << 3);
#pragma unroll
      for (int i = 0; i < 8; ++i) {
        int r = r0 + i;
        unsigned wd = adjc[r * 4 + (jc >> 5)];
        float bias = ((wd >> (jc & 31)) & 1u) ? qe1s[r] : qe0s[r];
        simf[r][jc] = (acc[i] + bias) * 0.125f;  // D^-0.5 = 1/8
      }
    }
    __syncthreads();

    // ---- softmax per row; a1 = sum(att*adj); pack att as f16 in place ----
    for (int r = w; r < MM; r += 8) {
      int c0 = lane * 4;
      float4 v = *(const float4*)&simf[r][c0];
      float mx = wave_max(fmaxf(fmaxf(v.x, v.y), fmaxf(v.z, v.w)));
      float ex = __expf(v.x - mx), ey = __expf(v.y - mx);
      float ez = __expf(v.z - mx), ew = __expf(v.w - mx);
      float sum = wave_sum(ex + ey + ez + ew);
      unsigned wd = adjc[r * 4 + (c0 >> 5)];
      int sb = c0 & 31;
      float a1p = (float)((wd >> (sb + 0)) & 1u) * ex +
                  (float)((wd >> (sb + 1)) & 1u) * ey +
                  (float)((wd >> (sb + 2)) & 1u) * ez +
                  (float)((wd >> (sb + 3)) & 1u) * ew;
      float a1 = wave_sum(a1p);
      float isum = 1.f / sum;
      if (lane == 0) a1s[r] = a1 * isum;
      _Float16* arow = (_Float16*)&simf[r][0];  // packed: row stride 256 halves
      arow[c0 + 0] = (_Float16)(ex * isum);
      arow[c0 + 1] = (_Float16)(ey * isum);
      arow[c0 + 2] = (_Float16)(ez * isum);
      arow[c0 + 3] = (_Float16)(ew * isum);
    }
    __syncthreads();

    // ---- out_h = att @ v + a1*e1 + (1-a1)*e0 ----
    const _Float16* att = (const _Float16*)&simf[0][0];
#pragma unroll
    for (int t = 0; t < 4; ++t) {
      int idx = w * 4 + t, rb = idx >> 2, cb = idx & 3;
      v8f acc = {};
#pragma unroll
      for (int kk = 0; kk < 4; ++kk) {
        v16h a = load_a(att + rb * 16 * 256 + kk * 32, 256);
        v16h b = load_b(&vt[cb * 16][kk * 32], 136);  // Bt[n=d][k=j] = v^T
        acc = wmma16(a, b, acc);
      }
      int dc = cb * 16 + (lane & 15);
      int r0 = rb * 16 + ((lane >> 4) << 3);
      float ev1 = e1[dc], ev0 = e0[dc];
#pragma unroll
      for (int i = 0; i < 8; ++i) {
        float a1v = a1s[r0 + i];
        oh[r0 + i][dc] = (_Float16)(acc[i] + a1v * ev1 + (1.f - a1v) * ev0);
      }
    }
    __syncthreads();

    // ---- accumulate Wo projection into persistent register tiles ----
#pragma unroll
    for (int kk = 0; kk < 2; ++kk) {
      v16h a = load_a(&oh[w * 16][kk * 32], 72);
#pragma unroll
      for (int cb = 0; cb < 8; ++cb) {
        v16h b = load_b(WoT + (size_t)(cb * 16) * INNR + h * DD + kk * 32, INNR);
        accP[cb] = wmma16(a, b, accP[cb]);
      }
    }
    __syncthreads();
  }  // head loop

  // ---- xn = proj + bo  (reuse simf as f32 xn buffer) ----
#pragma unroll
  for (int cb = 0; cb < 8; ++cb) {
    int col = cb * 16 + (lane & 15);
    int r0 = w * 16 + ((lane >> 4) << 3);
    float bb = bo[col];
#pragma unroll
    for (int i = 0; i < 8; ++i) simf[r0 + i][col] = accP[cb][i] + bb;
  }
  __syncthreads();

  // ---- gated residual 1 -> nodes1 ----
  for (int r = w; r < MM; r += 8) {
    int c0 = lane * 4;
    float4 xn = *(const float4*)&simf[r][c0];
    float4 rs = *(const float4*)&xres[r][c0];
    float dot = xn.x * g1[c0 + 0] + rs.x * g1[CC + c0 + 0] + (xn.x - rs.x) * g1[2 * CC + c0 + 0]
              + xn.y * g1[c0 + 1] + rs.y * g1[CC + c0 + 1] + (xn.y - rs.y) * g1[2 * CC + c0 + 1]
              + xn.z * g1[c0 + 2] + rs.z * g1[CC + c0 + 2] + (xn.z - rs.z) * g1[2 * CC + c0 + 2]
              + xn.w * g1[c0 + 3] + rs.w * g1[CC + c0 + 3] + (xn.w - rs.w) * g1[2 * CC + c0 + 3];
    dot = wave_sum(dot);
    float g = 1.f / (1.f + __expf(-dot));
    float4 o;
    o.x = xn.x * g + rs.x * (1.f - g);
    o.y = xn.y * g + rs.y * (1.f - g);
    o.z = xn.z * g + rs.z * (1.f - g);
    o.w = xn.w * g + rs.w * (1.f - g);
    *(float4*)&nodes1[((size_t)cl * MM + r) * CC + c0] = o;
  }
}

// ---------------- block 2: LN2 + FFN(gelu) + gate2 + final add ----------------

__global__ void __launch_bounds__(256)
k_block2(const float* __restrict__ nodes1, const float* __restrict__ xorig,
         const _Float16* __restrict__ W1T, const _Float16* __restrict__ W2T,
         const float* __restrict__ b1, const float* __restrict__ b2,
         const float* __restrict__ g2, const float* __restrict__ ln2g,
         const float* __restrict__ ln2b, float* __restrict__ out) {
  __shared__ float    xres[MM][CC];    // residual (nodes1)     64 KB
  __shared__ _Float16 y16[MM][136];    // LN2 output            34 KB
  __shared__ _Float16 h16[MM][520];    // gelu(y@W1+b1)        130 KB
  __shared__ float    xn[MM][CC];      // y2@W2+b2              64 KB

  const int tid = threadIdx.x, w = tid >> 5, lane = tid & 31;
  const int cl = blockIdx.x;

  tile_copy_in(nodes1 + (size_t)cl * MM * CC, &xres[0][0], tid);
  __syncthreads();

  for (int r = w; r < MM; r += 8) {
    const int c = lane * 4;
    float4 v = *(const float4*)&xres[r][c];
    float s  = wave_sum(v.x + v.y + v.z + v.w);
    float ss = wave_sum(v.x * v.x + v.y * v.y + v.z * v.z + v.w * v.w);
    float mu = s * (1.f / CC);
    float rs = rsqrtf(ss * (1.f / CC) - mu * mu + 1e-5f);
    y16[r][c + 0] = (_Float16)((v.x - mu) * rs * ln2g[c + 0] + ln2b[c + 0]);
    y16[r][c + 1] = (_Float16)((v.y - mu) * rs * ln2g[c + 1] + ln2b[c + 1]);
    y16[r][c + 2] = (_Float16)((v.z - mu) * rs * ln2g[c + 2] + ln2b[c + 2]);
    y16[r][c + 3] = (_Float16)((v.w - mu) * rs * ln2g[c + 3] + ln2b[c + 3]);
  }
  __syncthreads();

  // ---- h = gelu(y @ W1 + b1), [128,512] ----
  for (int t = 0; t < 32; ++t) {
    int idx = w * 32 + t, rb = idx >> 5, cb = idx & 31;
    v8f acc = {};
#pragma unroll
    for (int kk = 0; kk < 4; ++kk) {
      v16h a = load_a(&y16[rb * 16][kk * 32], 136);
      v16h b = load_b(W1T + (size_t)(cb * 16) * CC + kk * 32, CC);
      acc = wmma16(a, b, acc);
    }
    int col = cb * 16 + (lane & 15);
    int r0 = rb * 16 + ((lane >> 4) << 3);
    float bb = b1[col];
#pragma unroll
    for (int i = 0; i < 8; ++i) {
      float vv = acc[i] + bb;
      h16[r0 + i][col] = (_Float16)(0.5f * vv * (1.f + erff(vv * 0.70710678f)));
    }
  }
  __syncthreads();

  // ---- xn = h @ W2 + b2, [128,128]; wave w owns rows [16w,16w+16) ----
  {
    v8f acc2[8];
    v8f z = {};
#pragma unroll
    for (int i = 0; i < 8; ++i) acc2[i] = z;
    for (int kk = 0; kk < 16; ++kk) {
      v16h a = load_a(&h16[w * 16][kk * 32], 520);
#pragma unroll
      for (int cb = 0; cb < 8; ++cb) {
        v16h b = load_b(W2T + (size_t)(cb * 16) * INNR + kk * 32, INNR);
        acc2[cb] = wmma16(a, b, acc2[cb]);
      }
    }
#pragma unroll
    for (int cb = 0; cb < 8; ++cb) {
      int col = cb * 16 + (lane & 15);
      int r0 = w * 16 + ((lane >> 4) << 3);
      float bb = b2[col];
#pragma unroll
      for (int i = 0; i < 8; ++i) xn[r0 + i][col] = acc2[cb][i] + bb;
    }
  }
  __syncthreads();

  // ---- gated residual 2, then out = x_orig + nodes ----
  for (int r = w; r < MM; r += 8) {
    int c0 = lane * 4;
    float4 f = *(const float4*)&xn[r][c0];
    float4 rs = *(const float4*)&xres[r][c0];
    float dot = f.x * g2[c0 + 0] + rs.x * g2[CC + c0 + 0] + (f.x - rs.x) * g2[2 * CC + c0 + 0]
              + f.y * g2[c0 + 1] + rs.y * g2[CC + c0 + 1] + (f.y - rs.y) * g2[2 * CC + c0 + 1]
              + f.z * g2[c0 + 2] + rs.z * g2[CC + c0 + 2] + (f.z - rs.z) * g2[2 * CC + c0 + 2]
              + f.w * g2[c0 + 3] + rs.w * g2[CC + c0 + 3] + (f.w - rs.w) * g2[2 * CC + c0 + 3];
    dot = wave_sum(dot);
    float g = 1.f / (1.f + __expf(-dot));
    float4 xo = *(const float4*)(xorig + ((size_t)cl * MM + r) * CC + c0);
    float4 o;
    o.x = xo.x + f.x * g + rs.x * (1.f - g);
    o.y = xo.y + f.y * g + rs.y * (1.f - g);
    o.z = xo.z + f.z * g + rs.z * (1.f - g);
    o.w = xo.w + f.w * g + rs.w * (1.f - g);
    *(float4*)&out[((size_t)cl * MM + r) * CC + c0] = o;
  }
}

// ---------------- launcher ----------------

extern "C" void kernel_launch(void* const* d_in, const int* in_sizes, int n_in,
                              void* d_out, int out_size, void* d_ws, size_t ws_size,
                              hipStream_t stream) {
  const float* x       = (const float*)d_in[0];
  const int*   edge    = (const int*)d_in[1];
  // d_in[2] = clusters: arange(N).reshape(K,M) -> implicit contiguous blocking
  const float* adj_emb = (const float*)d_in[3];
  const float* ln1g    = (const float*)d_in[4];
  const float* ln1b    = (const float*)d_in[5];
  const float* Wq      = (const float*)d_in[6];
  const float* bq      = (const float*)d_in[7];
  const float* Wkv     = (const float*)d_in[8];
  const float* bkv     = (const float*)d_in[9];
  const float* We      = (const float*)d_in[10];
  const float* be      = (const float*)d_in[11];
  const float* Wo      = (const float*)d_in[12];
  const float* bo      = (const float*)d_in[13];
  const float* g1      = (const float*)d_in[14];
  const float* ln2g    = (const float*)d_in[15];
  const float* ln2b    = (const float*)d_in[16];
  const float* W1      = (const float*)d_in[17];
  const float* b1      = (const float*)d_in[18];
  const float* W2      = (const float*)d_in[19];
  const float* b2      = (const float*)d_in[20];
  const float* g2      = (const float*)d_in[21];

  char* ws = (char*)d_ws;
  // workspace layout (bytes)
  unsigned* adjw   = (unsigned*)(ws + 0);            //  64 KB
  float*    e01    = (float*)(ws + 65536);           //   4 KB
  _Float16* WqT    = (_Float16*)(ws + 69632);        // 128 KB
  _Float16* WkvT   = (_Float16*)(ws + 200704);       // 256 KB
  _Float16* WoT    = (_Float16*)(ws + 462848);       // 128 KB
  _Float16* W1T    = (_Float16*)(ws + 593920);       // 128 KB
  _Float16* W2T    = (_Float16*)(ws + 724992);       // 128 KB
  float*    nodes1 = (float*)(ws + 856064);          //   2 MB

  k_zero_adj<<<(KCL * MM * 4 + 255) / 256, 256, 0, stream>>>(adjw);
  k_build_adj<<<EE / 256, 256, 0, stream>>>(edge, adjw);
  k_prep<<<1540, 256, 0, stream>>>(Wq, Wkv, Wo, W1, W2, adj_emb, We, be,
                                   WqT, WkvT, WoT, W1T, W2T, e01);
  k_block1<<<KCL, 256, 0, stream>>>(x, adjw, e01, WqT, WkvT, WoT, bq, bkv, bo,
                                    g1, ln1g, ln1b, nodes1);
  k_block2<<<KCL, 256, 0, stream>>>(nodes1, x, W1T, W2T, b1, b2, g2, ln2g, ln2b,
                                    (float*)d_out);
}
